// CRF_32985348833704
// MI455X (gfx1250) — compile-verified, matrix operand order checked by reading, not run on hardware
//
#include <hip/hip_runtime.h>

// CRF log-likelihood for MI455X (gfx1250, wave32).
// B=256, S=512, K=128. Forward recursion as exp-domain GEMM per step via
// v_wmma_f32_16x16x32_f16. exp(transitions) f16 B-fragments live in VGPRs for
// all 511 steps; alpha state lives in registers; emissions triple-buffered
// into LDS via global_load_async_to_lds_b128 (ASYNCcnt), consumed two steps
// after issue. Manual split workgroup barriers (s_barrier_signal/-wait -1)
// with async issue + mask load + thread-local tile loads in the shadows.

#define S_LEN 512
#define KTAG  128
#define BTOT  256

#define AF_PITCH 136   // halves  (Af row pitch, pad for bank spread)
#define F_PITCH  132   // floats  (P / emission tile row pitch)

typedef __attribute__((ext_vector_type(16))) _Float16 v16h;
typedef __attribute__((ext_vector_type(8)))  float    v8f;

union Frag16 {
    v16h   h;
    float4 f4[2];
};

#define LOG2E 1.4426950408889634f
#define LN2   0.6931471805599453f

__device__ inline float raw_exp2(float x) {
#if __has_builtin(__builtin_amdgcn_exp2f)
    return __builtin_amdgcn_exp2f(x);   // v_exp_f32
#else
    return __expf(x * LN2);
#endif
}
__device__ inline float raw_log2(float x) {
#if __has_builtin(__builtin_amdgcn_logf)
    return __builtin_amdgcn_logf(x);    // v_log_f32
#else
    return __logf(x) * LOG2E;
#endif
}

// 32-bit LDS offset of a __shared__ pointer (generic addr low 32 bits)
__device__ inline unsigned lds_off(const void* p) {
    return (unsigned)(uintptr_t)p;
}

__device__ inline void barrier_signal() {  // flush LDS stores, then signal
    asm volatile("s_wait_dscnt 0x0\n\ts_barrier_signal -1" ::: "memory");
}
__device__ inline void barrier_wait() {
    asm volatile("s_barrier_wait -1" ::: "memory");
}

// ---------------------------------------------------------------------------
// expT_T[n*K + i] = exp(transitions[i*K + n])  (transposed, f16) so WMMA
// B-fragments (column n, K contiguous) are two b128 loads each.
// ---------------------------------------------------------------------------
__global__ void crf_expT_kernel(const float* __restrict__ trans,
                                _Float16* __restrict__ expT_T) {
    const int n = blockIdx.x;
    const int i = threadIdx.x;
    expT_T[n * KTAG + i] = (_Float16)raw_exp2(trans[i * KTAG + n] * LOG2E);
}

// ---------------------------------------------------------------------------
// Numerator score: one block per batch, 256 threads, tree-reduce in LDS.
// ---------------------------------------------------------------------------
__global__ __launch_bounds__(256) void crf_score_kernel(
    const float* __restrict__ em, const long long* __restrict__ tags,
    const unsigned char* __restrict__ mask,
    const float* __restrict__ startT, const float* __restrict__ endT,
    const float* __restrict__ trans, float* __restrict__ score) {
    const int b   = blockIdx.x;
    const int tid = threadIdx.x;
    __shared__ float sred[256];
    __shared__ int   cred[256];

    float part = 0.0f;
    int   cnt  = 0;
    for (int t = tid; t < S_LEN; t += 256) {
        const bool mt = mask[(size_t)b * S_LEN + t] != 0;
        cnt += mt ? 1 : 0;
        const int tg = mt ? (int)tags[(size_t)b * S_LEN + t] : 0;
        const float e = em[((size_t)b * S_LEN + t) * KTAG + tg];
        if (t == 0) {
            part += startT[tg] + e;  // t=0 term unconditional in reference
        } else {
            const bool mp = mask[(size_t)b * S_LEN + t - 1] != 0;
            const int  tp = mp ? (int)tags[(size_t)b * S_LEN + t - 1] : 0;
            part += mt ? (trans[tp * KTAG + tg] + e) : 0.0f;
        }
    }
    sred[tid] = part;
    cred[tid] = cnt;
    __syncthreads();
    for (int off = 128; off >= 1; off >>= 1) {
        if (tid < off) {
            sred[tid] += sred[tid + off];
            cred[tid] += cred[tid + off];
        }
        __syncthreads();
    }
    if (tid == 0) {
        int se = cred[0] - 1;
        if (se < 0) se = 0;
        const int lt = (mask[(size_t)b * S_LEN + se] != 0)
                           ? (int)tags[(size_t)b * S_LEN + se] : 0;
        score[b] = sred[0] + endT[lt];
    }
}

// ---------------------------------------------------------------------------
// Forward recursion. One workgroup = 16 batches, 8 waves; wave w owns output
// columns [16w,16w+16).
// ---------------------------------------------------------------------------
__global__ __launch_bounds__(256) void crf_forward_kernel(
    const float* __restrict__ em_g, const unsigned char* __restrict__ mask,
    const float* __restrict__ startT, const float* __restrict__ endT,
    const _Float16* __restrict__ expT_T, const float* __restrict__ score,
    float* __restrict__ out) {
    __shared__ alignas(16) _Float16 Af[16][AF_PITCH];     // exp(a-m), f16
    __shared__ alignas(16) float    P[16][F_PITCH];       // log(acc)+m
    __shared__ alignas(16) float    emt[3][16][F_PITCH];  // triple-buffered
    __shared__ alignas(16) float    mred[16];

    const int tid  = threadIdx.x;
    const int wave = tid >> 5;
    const int lane = tid & 31;
    const int b0   = blockIdx.x * 16;

    // cooperative mapping: 16 threads per batch row, 8 elems per thread
    const int bl = tid >> 4;
    const int g  = tid & 15;
    const int j0 = g * 8;

    // WMMA lane mapping (ISA 7.12.2)
    const int mrow = lane & 15;           // A row / D column index
    const int khlf = (lane >> 4) << 3;    // K (and D row) sub-offset per half
    const int ncol = (wave << 4) + mrow;  // this lane's N column

    const size_t brow = (size_t)(b0 + bl) * S_LEN;  // this thread's batch row

    // --- B fragments (exp of transitions): loaded once, resident in VGPRs.
    Frag16 bfr[4];
#pragma unroll
    for (int kc = 0; kc < 4; ++kc) {
        const _Float16* p = expT_T + (size_t)ncol * KTAG + kc * 32 + khlf;
        bfr[kc].f4[0] = *reinterpret_cast<const float4*>(p);
        bfr[kc].f4[1] = *reinterpret_cast<const float4*>(p + 16);
    }

    // rotating per-thread slot pointers into the three emission buffers:
    // erd = tile being consumed, ewr = tile being filled (2 steps ahead)
    float* erd = &emt[0][bl][j0];
    float* emd = &emt[1][bl][j0];
    float* ewr = &emt[2][bl][j0];

    // --- async-stage emissions tile t=1 into buffer 1 (32B per thread; each
    //     thread loads exactly the region it will itself consume later).
    {
        const float* src = em_g + (brow + 1) * KTAG + j0;
        const unsigned d = lds_off(emd);
        asm volatile("global_load_async_to_lds_b128 %0, %1, off"
                     :: "v"(d), "v"(src) : "memory");
        asm volatile("global_load_async_to_lds_b128 %0, %1, off offset:16"
                     :: "v"(d), "v"(src) : "memory");
    }

    // --- alpha_0 = start + emissions[:,0]  (held in registers)
    float av[8];
    {
        const float* src = em_g + brow * KTAG + j0;
#pragma unroll
        for (int q = 0; q < 8; ++q) av[q] = startT[j0 + q] + src[q];
    }
    unsigned char mkcur = 1;
    const float* gsrc = em_g + (brow + 2) * KTAG + j0;  // tile t+1 at t=1

    // dummy signal so every iteration strictly alternates signal/wait
    asm volatile("s_barrier_signal -1" ::: "memory");

    for (int t = 1; t < S_LEN; ++t) {
        // thread-local: tile t-1 resident after asynccnt wait (no barrier)
        float4 e0, e1;
        if (t >= 2) {
            asm volatile("s_wait_asynccnt 0x2" ::: "memory");
            e0 = *reinterpret_cast<const float4*>(erd);
            e1 = *reinterpret_cast<const float4*>(erd + 4);
        }
        barrier_wait();  // completes barrier B of step t-1: P now visible

        // resolve alpha_{t-1} = mask ? P_{t-1} + em_{t-1} : alpha_{t-2}
        if (t >= 2) {
            float4 p0 = *reinterpret_cast<const float4*>(&P[bl][j0]);
            float4 p1 = *reinterpret_cast<const float4*>(&P[bl][j0 + 4]);
            float nw[8] = {p0.x + e0.x, p0.y + e0.y, p0.z + e0.z, p0.w + e0.w,
                           p1.x + e1.x, p1.y + e1.y, p1.z + e1.z, p1.w + e1.w};
            const bool on = mkcur != 0;
#pragma unroll
            for (int q = 0; q < 8; ++q) av[q] = on ? nw[q] : av[q];
        }

        // per-batch max (butterfly leaves max in all 16 lanes of the group)
        float mv = av[0];
#pragma unroll
        for (int q = 1; q < 8; ++q) mv = fmaxf(mv, av[q]);
#pragma unroll
        for (int off = 8; off >= 1; off >>= 1)
            mv = fmaxf(mv, __shfl_xor(mv, off, 16));
        if (g == 0) mred[bl] = mv;

        // Af = exp(alpha - m) in f16
#pragma unroll
        for (int q = 0; q < 8; ++q)
            Af[bl][j0 + q] = (_Float16)raw_exp2((av[q] - mv) * LOG2E);

        barrier_signal();  // barrier A: Af published
        // --- barrier-A shadow: independent global-side work ---
        if (t < S_LEN - 1) {
            const unsigned d = lds_off(ewr);
            asm volatile("global_load_async_to_lds_b128 %0, %1, off"
                         :: "v"(d), "v"(gsrc) : "memory");
            asm volatile("global_load_async_to_lds_b128 %0, %1, off offset:16"
                         :: "v"(d), "v"(gsrc) : "memory");
            if (t < S_LEN - 2) __builtin_prefetch(gsrc + KTAG, 0, 0);
        }
        const unsigned char mknext = mask[brow + t];  // used at step t+1
        barrier_wait();

        // acc(16x16) = Af(16x128) * expT(128 x ntile): 8 b128 loads, 4 WMMAs
        Frag16 afr[4];
#pragma unroll
        for (int kc = 0; kc < 4; ++kc) {
            const _Float16* ap = &Af[mrow][kc * 32 + khlf];
            afr[kc].f4[0] = *reinterpret_cast<const float4*>(ap);
            afr[kc].f4[1] = *reinterpret_cast<const float4*>(ap + 16);
        }
        v8f acc = {};
#pragma unroll
        for (int kc = 0; kc < 4; ++kc)
            acc = __builtin_amdgcn_wmma_f32_16x16x32_f16(
                false, afr[kc].h, false, bfr[kc].h, (short)0, acc, false,
                false);

        // writeback pre-emission score P = log(acc) + m  (unconditional)
        float mr[8];
        {
            float4 m0 = *reinterpret_cast<const float4*>(&mred[khlf]);
            float4 m1 = *reinterpret_cast<const float4*>(&mred[khlf + 4]);
            mr[0] = m0.x; mr[1] = m0.y; mr[2] = m0.z; mr[3] = m0.w;
            mr[4] = m1.x; mr[5] = m1.y; mr[6] = m1.z; mr[7] = m1.w;
        }
#pragma unroll
        for (int v = 0; v < 8; ++v)
            P[khlf + v][ncol] = raw_log2(acc[v]) * LN2 + mr[v];

        barrier_signal();  // barrier B: P published (waited next iteration)

        // rotate buffers / advance pipelined state
        float* tmp = erd; erd = emd; emd = ewr; ewr = tmp;
        mkcur = mknext;
        gsrc += KTAG;
    }

    // final resolve (step 511) then logZ = logsumexp(alpha + end)
    {
        asm volatile("s_wait_asynccnt 0x0" ::: "memory");
        float4 e0 = *reinterpret_cast<const float4*>(erd);
        float4 e1 = *reinterpret_cast<const float4*>(erd + 4);
        barrier_wait();  // matches last barrier-B signal
        float4 p0 = *reinterpret_cast<const float4*>(&P[bl][j0]);
        float4 p1 = *reinterpret_cast<const float4*>(&P[bl][j0 + 4]);
        float nw[8] = {p0.x + e0.x, p0.y + e0.y, p0.z + e0.z, p0.w + e0.w,
                       p1.x + e1.x, p1.y + e1.y, p1.z + e1.z, p1.w + e1.w};
        const bool on = mkcur != 0;
#pragma unroll
        for (int q = 0; q < 8; ++q) av[q] = on ? nw[q] : av[q];

        float vals[8];
        float vmax = -3.4e38f;
#pragma unroll
        for (int q = 0; q < 8; ++q) {
            vals[q] = av[q] + endT[j0 + q];
            vmax = fmaxf(vmax, vals[q]);
        }
#pragma unroll
        for (int off = 8; off >= 1; off >>= 1)
            vmax = fmaxf(vmax, __shfl_xor(vmax, off, 16));
        float s = 0.0f;
#pragma unroll
        for (int q = 0; q < 8; ++q) s += raw_exp2((vals[q] - vmax) * LOG2E);
#pragma unroll
        for (int off = 8; off >= 1; off >>= 1) s += __shfl_xor(s, off, 16);
        if (g == 0)
            out[b0 + bl] = score[b0 + bl] - (vmax + raw_log2(s) * LN2);
    }
}

// ---------------------------------------------------------------------------
extern "C" void kernel_launch(void* const* d_in, const int* in_sizes, int n_in,
                              void* d_out, int out_size, void* d_ws,
                              size_t ws_size, hipStream_t stream) {
    const float*         emissions = (const float*)d_in[0];
    const long long*     tags      = (const long long*)d_in[1];    // int64
    const unsigned char* mask      = (const unsigned char*)d_in[2]; // bool
    const float*         startT    = (const float*)d_in[3];
    const float*         endT      = (const float*)d_in[4];
    const float*         trans     = (const float*)d_in[5];
    float*               out       = (float*)d_out;

    _Float16* expT_T = (_Float16*)d_ws;  // 32 KB
    float* score =
        (float*)((char*)d_ws + (size_t)KTAG * KTAG * sizeof(_Float16));

    crf_expT_kernel<<<KTAG, KTAG, 0, stream>>>(trans, expT_T);
    crf_score_kernel<<<BTOT, 256, 0, stream>>>(emissions, tags, mask, startT,
                                               endT, trans, score);
    crf_forward_kernel<<<BTOT / 16, 256, 0, stream>>>(
        emissions, mask, startT, endT, expT_T, score, out);
}
